// AttentionPartition_6253472383797
// MI455X (gfx1250) — matrix operand order repaired and checked by reading, not compiled, not run on hardware
//
#include <hip/hip_runtime.h>

// ---------------- problem constants ----------------
#define NB   32
#define LT   1568
#define DT   768
#define HDS  12
#define DH   64
#define PARTS 49
// GEMM rows: N*L = 50176 = 784 * 64

typedef __attribute__((ext_vector_type(16))) __bf16 bf16x16;
typedef __attribute__((ext_vector_type(8)))  __bf16 bf16x8;
typedef __attribute__((ext_vector_type(8)))  float  f32x8;
typedef __attribute__((ext_vector_type(4)))  float  f32x4;
typedef __attribute__((ext_vector_type(4)))  unsigned short u16x4;
typedef __attribute__((ext_vector_type(8)))  unsigned short u16x8;
typedef __attribute__((ext_vector_type(4)))  unsigned int   u32x4;
typedef __attribute__((ext_vector_type(8)))  int            i32x8;
typedef __attribute__((ext_vector_type(4)))  int            i32x4;

// fp32 -> bf16 hi (truncate, residual exact) + bf16 lo (RNE). a ~= hi+lo, |err| ~ 2^-17|a|.
__device__ __forceinline__ void split2bf16(float f, unsigned short& h, unsigned short& l) {
  unsigned u = __builtin_bit_cast(unsigned, f);
  h = (unsigned short)(u >> 16);
  float fh = __builtin_bit_cast(float, u & 0xFFFF0000u);
  float fl = f - fh;
  unsigned ul = __builtin_bit_cast(unsigned, fl);
  unsigned r = ul + 0x7FFFu + ((ul >> 16) & 1u);
  l = (unsigned short)(r >> 16);
}

// WMMA A operand (16x32 bf16): element e -> K = (e/8)*16 + half*8 + e%8
__device__ __forceinline__ bf16x16 ldsA(const unsigned short* p, int stride, int row, int k0, int half) {
  const __bf16* b = (const __bf16*)p + (size_t)row * stride + k0 + half * 8;
  bf16x8 c0 = *(const bf16x8*)(b);
  bf16x8 c1 = *(const bf16x8*)(b + 16);
  return __builtin_shufflevector(c0, c1, 0,1,2,3,4,5,6,7,8,9,10,11,12,13,14,15);
}
// WMMA B operand (32x16 bf16): element e -> K = (e/8)*16 + half*8 + e%8
__device__ __forceinline__ bf16x16 ldsB(const unsigned short* p, int stride, int row, int k0, int half) {
  const __bf16* b = (const __bf16*)p + (size_t)row * stride + k0 + half * 8;
  bf16x8 c0 = *(const bf16x8*)(b);
  bf16x8 c1 = *(const bf16x8*)(b + 16);
  return __builtin_shufflevector(c0, c1, 0,1,2,3,4,5,6,7,8,9,10,11,12,13,14,15);
}
__device__ __forceinline__ f32x8 wmma_bf16(bf16x16 a, bf16x16 b, f32x8 c) {
  return __builtin_amdgcn_wmma_f32_16x16x32_bf16(false, a, false, b, (short)0, c, false, false);
}
// split-bf16: a*b ~= ah*bh + ah*bl + al*bh (3 WMMAs, near-fp32 result)
__device__ __forceinline__ f32x8 mma_split(
    const unsigned short* Ah, const unsigned short* Al, int lda, int arow,
    const unsigned short* Bh, const unsigned short* Bl, int ldb, int brow,
    int k0, int half, f32x8 acc) {
  bf16x16 ah = ldsA(Ah, lda, arow, k0, half);
  bf16x16 al = ldsA(Al, lda, arow, k0, half);
  bf16x16 bh = ldsB(Bh, ldb, brow, k0, half);
  bf16x16 bl = ldsB(Bl, ldb, brow, k0, half);
  acc = wmma_bf16(ah, bh, acc);
  acc = wmma_bf16(ah, bl, acc);
  acc = wmma_bf16(al, bh, acc);
  return acc;
}

// Generic pointer -> LDS byte offset (LDS aperture: offset is the low 32 bits of the address, ISA 10.2)
__device__ __forceinline__ unsigned lds_byte_off(const void* p) {
  return (unsigned)(uintptr_t)p;
}

// ---------------------------------------------------------------------------
// TDM: DMA a 2D u16 tile [tileRows x 32] from a hi/lo plane (row length 768)
// into LDS with hardware padding to a 48-u16 row stride (pad 8 DWORDs after
// every 16 DWORDs). D# per CDNA5 ISA 8.3-8.6; trailing groups zero (2D tensor,
// no gather, not in a cluster). This toolchain uses the 6-arg builtin form.
// ---------------------------------------------------------------------------
__device__ __forceinline__ void tdm_load_tile(unsigned ldsOff, const unsigned short* plane,
                                              unsigned row0, unsigned k0,
                                              unsigned tensorRows, unsigned tileRows) {
  unsigned long long ga = (unsigned long long)(uintptr_t)plane +
                          2ull * ((unsigned long long)row0 * (unsigned long long)DT + k0);
  u32x4 g0;
  g0[0] = 1u;                                   // count=1 (valid), user mode
  g0[1] = ldsOff;                               // lds_addr (bytes)
  g0[2] = (unsigned)ga;                         // global_addr[31:0]
  g0[3] = (unsigned)((ga >> 32) & 0x01FFFFFFull) | (2u << 30); // addr[56:32] | type=2
  i32x8 g1;
  g1[0] = (int)((1u << 16)        // data_size = 2 bytes
              | (1u << 20)        // pad_enable
              | (3u << 22)        // pad_interval: 16 DWORDs (= 32 u16 = tile row)
              | (7u << 25));      // pad_amount: 8 DWORDs (= 16 u16) -> stride 48
  g1[1] = (int)(((unsigned)DT & 0xFFFFu) << 16);                 // tensor_dim0 lo16
  g1[2] = (int)(((unsigned)DT >> 16) | ((tensorRows & 0xFFFFu) << 16)); // dim0 hi | dim1 lo
  g1[3] = (int)((tensorRows >> 16) | (32u << 16));               // dim1 hi | tile_dim0=32
  g1[4] = (int)tileRows;                                         // tile_dim1 | tile_dim2=0
  g1[5] = DT;                                                    // tensor_dim0_stride lo32
  g1[6] = 0;
  g1[7] = 0;
  i32x4 z4 = {0, 0, 0, 0};
  i32x8 z8 = {0, 0, 0, 0, 0, 0, 0, 0};
  __builtin_amdgcn_tensor_load_to_lds(g0, g1, z4, z4, z8, 0);
}

// ---------------------------------------------------------------------------
// elementwise fp32 -> (hi, lo) bf16 planes
// ---------------------------------------------------------------------------
__global__ __launch_bounds__(256) void split_kernel(const float* __restrict__ src,
                                                    unsigned short* __restrict__ dh,
                                                    unsigned short* __restrict__ dl, int n4) {
  int i = blockIdx.x * 256 + threadIdx.x;
  if (i >= n4) return;
  f32x4 v = *(const f32x4*)(src + (size_t)i * 4);
  u16x4 vh, vl;
#pragma unroll
  for (int k = 0; k < 4; ++k) {
    unsigned short h, l;
    split2bf16(v[k], h, l);
    vh[k] = h; vl[k] = l;
  }
  *(u16x4*)(dh + (size_t)i * 4) = vh;
  *(u16x4*)(dl + (size_t)i * 4) = vl;
}

// ---------------------------------------------------------------------------
// C[m,n] = sum_k A[m,k]*W[n,k]. A/W given as pre-split bf16 hi/lo planes
// (row length 768). Block tile 64x64, K-step 32, 256 threads = 8 waves (4x2),
// TDM double-buffered LDS staging. Epilogue: hi/lo planes (col<splitN -> C0,
// else C1) or fp32 + bias when Cf != nullptr.
// ---------------------------------------------------------------------------
__global__ __launch_bounds__(256) void gemm_tdm_kernel(
    const unsigned short* __restrict__ Agh, const unsigned short* __restrict__ Agl,
    const unsigned short* __restrict__ Wgh, const unsigned short* __restrict__ Wgl,
    int aRows, int wRows,
    unsigned short* __restrict__ C0h, unsigned short* __restrict__ C0l,
    unsigned short* __restrict__ C1h, unsigned short* __restrict__ C1l,
    float* __restrict__ Cf, const float* __restrict__ bias, int splitN) {
  __shared__ __attribute__((aligned(32))) unsigned short Ab[2][2][64 * 48];
  __shared__ __attribute__((aligned(32))) unsigned short Wb[2][2][64 * 48];

  const int tid  = threadIdx.x;
  const int lane = tid & 31;
  const int wid  = tid >> 5;
  const int half = lane >> 4;
  const int ln   = lane & 15;
  const int waveM = wid >> 1;   // 0..3 -> 16-row slice
  const int waveN = wid & 1;    // 0..1 -> 32-col slice (2 tiles)
  const int m0 = blockIdx.y * 64;
  const int n0 = blockIdx.x * 64;

  f32x8 acc[2] = {};

  // prologue: fill buffer 0
  if (wid == 0) {
    tdm_load_tile(lds_byte_off(Ab[0][0]), Agh, m0, 0, (unsigned)aRows, 64);
    tdm_load_tile(lds_byte_off(Ab[0][1]), Agl, m0, 0, (unsigned)aRows, 64);
    tdm_load_tile(lds_byte_off(Wb[0][0]), Wgh, n0, 0, (unsigned)wRows, 64);
    tdm_load_tile(lds_byte_off(Wb[0][1]), Wgl, n0, 0, (unsigned)wRows, 64);
  }

  int cur = 0;
  const int KSTEPS = DT / 32;
  for (int it = 0; it < KSTEPS; ++it) {
    if (wid == 0) {
      if (it + 1 < KSTEPS) {  // prefetch next K-slice into the other buffer
        int k0 = (it + 1) * 32;
        tdm_load_tile(lds_byte_off(Ab[cur ^ 1][0]), Agh, m0, k0, (unsigned)aRows, 64);
        tdm_load_tile(lds_byte_off(Ab[cur ^ 1][1]), Agl, m0, k0, (unsigned)aRows, 64);
        tdm_load_tile(lds_byte_off(Wb[cur ^ 1][0]), Wgh, n0, k0, (unsigned)wRows, 64);
        tdm_load_tile(lds_byte_off(Wb[cur ^ 1][1]), Wgl, n0, k0, (unsigned)wRows, 64);
        __builtin_amdgcn_s_wait_tensorcnt((short)4);  // current buffer's 4 done
      } else {
        __builtin_amdgcn_s_wait_tensorcnt((short)0);  // drain
      }
    }
    __syncthreads();
#pragma unroll
    for (int j = 0; j < 2; ++j)
      acc[j] = mma_split(Ab[cur][0], Ab[cur][1], 48, waveM * 16 + ln,
                         Wb[cur][0], Wb[cur][1], 48, waveN * 32 + j * 16 + ln,
                         0, half, acc[j]);
    __syncthreads();   // all reads of buf done before wave0 re-fills it
    cur ^= 1;
  }

#pragma unroll
  for (int j = 0; j < 2; ++j) {
    int col = n0 + waveN * 32 + j * 16 + ln;
#pragma unroll
    for (int r = 0; r < 8; ++r) {
      int row = m0 + waveM * 16 + half * 8 + r;
      float v = acc[j][r];
      if (Cf) {
        Cf[(size_t)row * DT + col] = v + bias[col];
      } else {
        unsigned short h, l;
        split2bf16(v, h, l);
        if (col < splitN) {
          C0h[(size_t)row * DT + col] = h;
          C0l[(size_t)row * DT + col] = l;
        } else {
          C1h[(size_t)row * DT + col - splitN] = h;
          C1l[(size_t)row * DT + col - splitN] = l;
        }
      }
    }
  }
}

// ---------------------------------------------------------------------------
// Attention per (partition p, head h); faithful q=k bug -> S = Kg Kg^T * scale.
// K/V arrive as pre-split bf16 planes (no conversion here). Output written as
// hi/lo planes for the final TDM GEMM. 49 padded to 64 with explicit zeros.
// ---------------------------------------------------------------------------
__global__ __launch_bounds__(128) void attn_kernel(
    const unsigned short* __restrict__ Khi, const unsigned short* __restrict__ Klo,
    const unsigned short* __restrict__ Vhi, const unsigned short* __restrict__ Vlo,
    const int* __restrict__ ids,
    unsigned short* __restrict__ Ohi, unsigned short* __restrict__ Olo) {
  __shared__ __attribute__((aligned(32))) unsigned short KHh[64 * 80], KHl[64 * 80]; // K rows; reused for P
  __shared__ __attribute__((aligned(32))) unsigned short VTh[64 * 80], VTl[64 * 80]; // V transposed [d][k]
  __shared__ float Sb[64 * 68];

  const int tid  = threadIdx.x;
  const int lane = tid & 31;
  const int wid  = tid >> 5;   // 0..3
  const int half = lane >> 4;
  const int ln   = lane & 15;
  const int h = blockIdx.x;
  const int p = blockIdx.y;
  const int n = p >> 5;
  const int lbase = (p & 31) * PARTS;

  for (int e = tid; e < 64 * 80; e += 128) { KHh[e] = 0; KHl[e] = 0; VTh[e] = 0; VTl[e] = 0; }
  __syncthreads();

  // gather permuted rows for this head (pure u16 copies; V transposed)
  for (int e = tid; e < PARTS * 8; e += 128) {   // 49 rows x 8 chunks of 8 u16
    int j = e >> 3, c8 = (e & 7) * 8;
    int sid = ids[n * LT + lbase + j];
    size_t base = (size_t)(n * LT + sid) * DT + h * DH + c8;
    u16x8 kh = *(const u16x8*)(Khi + base);
    u16x8 kl = *(const u16x8*)(Klo + base);
    *(u16x8*)(&KHh[j * 80 + c8]) = kh;
    *(u16x8*)(&KHl[j * 80 + c8]) = kl;
    u16x8 vh = *(const u16x8*)(Vhi + base);
    u16x8 vl = *(const u16x8*)(Vlo + base);
#pragma unroll
    for (int i = 0; i < 8; ++i) {
      VTh[(c8 + i) * 80 + j] = vh[i];
      VTl[(c8 + i) * 80 + j] = vl[i];
    }
  }
  __syncthreads();

  // S = Kg Kg^T * 0.125 (4x4 tiles of 16x16; one tile column per wave)
  for (int t = wid; t < 16; t += 4) {
    int ti = t >> 2, tj = t & 3;
    f32x8 acc = {};
#pragma unroll
    for (int ks = 0; ks < 2; ++ks)
      acc = mma_split(KHh, KHl, 80, ti * 16 + ln,
                      KHh, KHl, 80, tj * 16 + ln, ks * 32, half, acc);
#pragma unroll
    for (int r = 0; r < 8; ++r)
      Sb[(ti * 16 + half * 8 + r) * 68 + tj * 16 + ln] = acc[r] * 0.125f;
  }
  __syncthreads();

  // row softmax over key axis; P (bf16 hi/lo) replaces K in LDS
  if (tid < 64) {
    int q = tid;
    if (q < PARTS) {
      float mx = -3.4e38f;
      for (int c = 0; c < PARTS; ++c) mx = fmaxf(mx, Sb[q * 68 + c]);
      float s = 0.f;
      for (int c = 0; c < PARTS; ++c) {
        float ev = __expf(Sb[q * 68 + c] - mx);
        Sb[q * 68 + c] = ev;
        s += ev;
      }
      float inv = 1.0f / s;
      for (int c = 0; c < PARTS; ++c) {
        unsigned short hh, ll;
        split2bf16(Sb[q * 68 + c] * inv, hh, ll);
        KHh[q * 80 + c] = hh; KHl[q * 80 + c] = ll;
      }
      for (int c = PARTS; c < 64; ++c) { KHh[q * 80 + c] = 0; KHl[q * 80 + c] = 0; }
    } else {
      for (int c = 0; c < 64; ++c) { KHh[q * 80 + c] = 0; KHl[q * 80 + c] = 0; }
    }
  }
  __syncthreads();

  // O = P @ V, emitted as hi/lo planes (still in shuffled order, as reference)
  for (int t = wid; t < 16; t += 4) {
    int ti = t >> 2, tj = t & 3;
    f32x8 acc = {};
#pragma unroll
    for (int ks = 0; ks < 2; ++ks)
      acc = mma_split(KHh, KHl, 80, ti * 16 + ln,
                      VTh, VTl, 80, tj * 16 + ln, ks * 32, half, acc);
#pragma unroll
    for (int r = 0; r < 8; ++r) {
      int q = ti * 16 + half * 8 + r;
      if (q < PARTS) {
        unsigned short hh, ll;
        split2bf16(acc[r], hh, ll);
        size_t o = (size_t)(p * PARTS + q) * DT + h * DH + tj * 16 + ln;
        Ohi[o] = hh; Olo[o] = ll;
      }
    }
  }
}

extern "C" void kernel_launch(void* const* d_in, const int* in_sizes, int n_in,
                              void* d_out, int out_size, void* d_ws, size_t ws_size,
                              hipStream_t stream) {
  (void)in_sizes; (void)n_in; (void)out_size; (void)ws_size;
  const float* x      = (const float*)d_in[0];
  const float* w_qkv  = (const float*)d_in[1];
  const float* w_proj = (const float*)d_in[2];
  const float* b_proj = (const float*)d_in[3];
  const int*   ids    = (const int*)d_in[4];
  float* out = (float*)d_out;

  const size_t MAT  = (size_t)NB * LT * DT;   // 38,535,168
  const size_t WKV  = (size_t)2 * DT * DT;    // 1,179,648
  const size_t WPJ  = (size_t)DT * DT;        // 589,824
  unsigned short* w = (unsigned short*)d_ws;
  unsigned short* Xh   = w;            w += MAT;
  unsigned short* Xl   = w;            w += MAT;
  unsigned short* Khi  = w;            w += MAT;
  unsigned short* Klo  = w;            w += MAT;
  unsigned short* Vhi  = w;            w += MAT;
  unsigned short* Vlo  = w;            w += MAT;
  unsigned short* Ohi  = w;            w += MAT;
  unsigned short* Olo  = w;            w += MAT;
  unsigned short* Wkvh = w;            w += WKV;
  unsigned short* Wkvl = w;            w += WKV;
  unsigned short* Wph  = w;            w += WPJ;
  unsigned short* Wpl  = w;            w += WPJ;

  // 0) pre-split fp32 -> bf16 hi/lo planes (same total bytes as fp32)
  split_kernel<<<(int)((MAT / 4 + 255) / 256), 256, 0, stream>>>(x, Xh, Xl, (int)(MAT / 4));
  split_kernel<<<(int)((WKV / 4 + 255) / 256), 256, 0, stream>>>(w_qkv + (size_t)DT * DT, Wkvh, Wkvl, (int)(WKV / 4));
  split_kernel<<<(int)((WPJ / 4 + 255) / 256), 256, 0, stream>>>(w_proj, Wph, Wpl, (int)(WPJ / 4));

  // 1) K,V projections (Q dead due to faithful q=k bug); TDM-staged GEMM.
  //    1536 output cols: col-blocks 0..11 -> K planes, 12..23 -> V planes.
  gemm_tdm_kernel<<<dim3(24, 784), 256, 0, stream>>>(
      Xh, Xl, Wkvh, Wkvl, NB * LT, 2 * DT,
      Khi, Klo, Vhi, Vlo, nullptr, nullptr, DT);

  // 2) partitioned attention (gather by shuffle_ids at load time)
  attn_kernel<<<dim3(HDS, 1024), 128, 0, stream>>>(Khi, Klo, Vhi, Vlo, ids, Ohi, Olo);

  // 3) output projection + bias -> d_out (fp32)
  gemm_tdm_kernel<<<dim3(12, 784), 256, 0, stream>>>(
      Ohi, Olo, Wph, Wpl, NB * LT, DT,
      nullptr, nullptr, nullptr, nullptr, out, b_proj, 1 << 30);
}
